// BiGRUStack_46007689675060
// MI455X (gfx1250) — matrix operand order repaired and checked by reading, not compiled, not run on hardware
//
#include <hip/hip_runtime.h>
#include <cstdint>

#define TT   4096
#define HH   1024
#define G3   3072          // 3*H
#define NBLK 48            // scan grid: 24 blocks x 2 directions
#define KCH  128           // K-columns staged per LDS round in the GEMM

typedef __attribute__((ext_vector_type(16))) __bf16 v16bf;
typedef __attribute__((ext_vector_type(8)))  float  v8f;

union Frag32B { uint4 u[2]; v16bf v; };

struct Bar { unsigned count; unsigned gen; };

static __device__ __forceinline__ float bflo(unsigned u) { return __uint_as_float(u << 16); }
static __device__ __forceinline__ float bfhi(unsigned u) { return __uint_as_float(u & 0xffff0000u); }
static __device__ __forceinline__ unsigned short f2bf(float f) {
  return __builtin_bit_cast(unsigned short, (__bf16)f);   // RNE convert
}

// ---------------------------------------------------------------- utilities
__global__ void init_bar_kernel(Bar* bar) {
  if (threadIdx.x == 0) { bar->count = 0u; bar->gen = 0u; }
}

__global__ void f32_to_bf16_kernel(const float* __restrict__ in,
                                   unsigned short* __restrict__ out, long n) {
  long i = (long)blockIdx.x * blockDim.x + threadIdx.x;
  if (i < n) out[i] = f2bf(in[i]);
}

// w_hh [3072 x 1024] f32  ->  packed bf16 [k/8][3072] uint4 (8 K-values per element)
// Per scan iteration a wave then loads 32 lanes x 16B = fully coalesced 512B.
__global__ void pack_whh_kernel(const float* __restrict__ w, uint4* __restrict__ out) {
  long i = (long)blockIdx.x * blockDim.x + threadIdx.x;
  if (i >= (long)(HH / 8) * G3) return;
  int r  = (int)(i % G3);
  int k8 = (int)(i / G3);
  const float* s = w + (long)r * HH + k8 * 8;
  uint4 o;
  o.x = (unsigned)f2bf(s[0]) | ((unsigned)f2bf(s[1]) << 16);
  o.y = (unsigned)f2bf(s[2]) | ((unsigned)f2bf(s[3]) << 16);
  o.z = (unsigned)f2bf(s[4]) | ((unsigned)f2bf(s[5]) << 16);
  o.w = (unsigned)f2bf(s[6]) | ((unsigned)f2bf(s[7]) << 16);
  out[i] = o;
}

// ---------------------------------------------------------------- WMMA GEMM
// out[T x 3072] = X[T x K](bf16) * W[3072 x K](bf16)^T + bias, fp32 accumulate.
// Block: 256 threads = 8 waves. Each block computes a 32(M) x 128(N) strip:
// each wave owns one 16-wide N-tile and TWO 16-row M-tiles that share a single
// B fragment (register-level B reuse). The 32x128 bf16 A panel is staged into
// 8 KB of LDS once per 128-K round: one global_load_b128 + one ds_store_b128
// per thread, 2 barriers per 128 K (4x fewer than per-32K staging).
// grid = (24 ntile-groups, T/32 mtiles, 2 directions).
__global__ __launch_bounds__(256, 2) void wmma_igates_kernel(
    const unsigned short* __restrict__ X,
    const unsigned short* __restrict__ Wf, const unsigned short* __restrict__ Wb,
    const float* __restrict__ biasF, const float* __restrict__ biasB,
    float* __restrict__ outF, float* __restrict__ outB, int K)
{
  const unsigned short* W = blockIdx.z ? Wb : Wf;
  const float* bias       = blockIdx.z ? biasB : biasF;
  float* out              = blockIdx.z ? outB : outF;

  __shared__ unsigned short sA[32 * KCH];   // 8 KB

  const int tid   = threadIdx.x;
  const int lane  = tid & 31;
  const int wave  = tid >> 5;
  const int mtile = blockIdx.y;                   // 32-row M tile
  const int ntile = blockIdx.x * 8 + wave;
  const int n     = ntile * 16 + (lane & 15);     // output column (gate index)
  const int kbA   = (lane >> 4) * 8;              // A fragment K-base per half-wave
  const int kbB   = (lane >> 4) * 8;              // B fragment K-base per half-wave
  const int srow  = tid >> 3;                     // staging row 0..31
  const int scol  = (tid & 7) * 16;               // staging col 0..112
  const int ar    = lane & 15;
  const long abase = (long)mtile * 32;

  v8f acc0 = {}, acc1 = {};
  for (int kk = 0; kk < K; kk += KCH) {
    __syncthreads();
    // stage 32x128 bf16 A panel: one b128 load + one b128 LDS store per thread
    *(uint4*)&sA[srow * KCH + scol] =
        *(const uint4*)&X[(abase + srow) * K + kk + scol];
    __syncthreads();

#pragma unroll
    for (int ks = 0; ks < KCH; ks += 32) {
      Frag32B a0, a1, b;
      // A 16x32 fragments: lane holds row (lane&15) [+16 for second M tile],
      // K = {kbA..kbA+7, kbA+16..kbA+23} within this 32-K step
      a0.u[0] = *(const uint4*)&sA[ar * KCH + ks + kbA];
      a0.u[1] = *(const uint4*)&sA[ar * KCH + ks + kbA + 16];
      a1.u[0] = *(const uint4*)&sA[(ar + 16) * KCH + ks + kbA];
      a1.u[1] = *(const uint4*)&sA[(ar + 16) * KCH + ks + kbA + 16];
      // B 32x16: lane holds col (lane&15), K = {kbB..kbB+7, kbB+16..kbB+23}
      const uint4* pb = (const uint4*)&W[(long)n * K + kk + ks + kbB];
      b.u[0] = pb[0];
      b.u[1] = pb[2];

      acc0 = __builtin_amdgcn_wmma_f32_16x16x32_bf16(false, a0.v, false, b.v,
                                                     (short)0, acc0, false, false);
      acc1 = __builtin_amdgcn_wmma_f32_16x16x32_bf16(false, a1.v, false, b.v,
                                                     (short)0, acc1, false, false);
    }
    if (kk + KCH < K) __builtin_prefetch(&W[(long)n * K + kk + KCH], 0, 1);
  }

  const float bv   = bias[n];
  const int   mrow = mtile * 32 + (lane >> 4) * 8;   // C/D: M = v + 8*(lane>=16)
#pragma unroll
  for (int v = 0; v < 8; ++v) {
    out[(long)(mrow + v) * G3 + n]      = acc0[v] + bv;
    out[(long)(mrow + 16 + v) * G3 + n] = acc1[v] + bv;
  }
}

// ---------------------------------------------------------------- grid barrier
static __device__ __forceinline__ void grid_barrier(Bar* bar, unsigned nblk) {
  __syncthreads();
  if (threadIdx.x == 0) {
    __threadfence();   // make prior global stores visible at agent scope
    unsigned g = __hip_atomic_load(&bar->gen, __ATOMIC_RELAXED, __HIP_MEMORY_SCOPE_AGENT);
    if (__hip_atomic_fetch_add(&bar->count, 1u, __ATOMIC_ACQ_REL,
                               __HIP_MEMORY_SCOPE_AGENT) == nblk - 1u) {
      __hip_atomic_store(&bar->count, 0u, __ATOMIC_RELAXED, __HIP_MEMORY_SCOPE_AGENT);
      __hip_atomic_store(&bar->gen, g + 1u, __ATOMIC_RELEASE, __HIP_MEMORY_SCOPE_AGENT);
    } else {
      while (__hip_atomic_load(&bar->gen, __ATOMIC_ACQUIRE,
                               __HIP_MEMORY_SCOPE_AGENT) == g) {
        __builtin_amdgcn_s_sleep(2);
      }
    }
    __threadfence();
  }
  __syncthreads();
}

// ---------------------------------------------------------------- GRU scan
// 48 blocks x 128 threads, resident/persistent. blocks [0,24) = fwd, [24,48) = bwd.
// One thread per hgates row (3072 per direction). h lives in a per-direction
// global buffer, staged into LDS each step. Two grid barriers per timestep.
__global__ __launch_bounds__(128, 1) void gru_scan_kernel(
    const float* __restrict__ igF, const float* __restrict__ igB,   // [T x 3072]
    const uint4* __restrict__ wqF, const uint4* __restrict__ wqB,   // packed w_hh
    const float* __restrict__ bnF, const float* __restrict__ bnB,   // [1024]
    unsigned short* __restrict__ xnext_bf,   // [T x 2048] bf16 (next-layer input)
    float* __restrict__ out_f32,             // [T x 2048] f32 (d_out on last layer)
    int write_f32,
    float* __restrict__ hbuf,                // [2 x 1024]
    float* __restrict__ hg,                  // [2 x 3072]
    Bar* bar)
{
  __shared__ float hs[HH];

  const int dir = blockIdx.x / 24;
  const int rb  = blockIdx.x % 24;
  const int r   = rb * 128 + threadIdx.x;    // 0..3071

  const float* ig = dir ? igB : igF;
  const uint4* wq = dir ? wqB : wqF;
  const float* bn = dir ? bnB : bnF;
  float* h   = hbuf + dir * HH;
  float* hgd = hg   + dir * G3;

  if (r < HH) h[r] = 0.0f;                   // h0 = 0
  grid_barrier(bar, NBLK);

  for (int step = 0; step < TT; ++step) {
    const int t = dir ? (TT - 1 - step) : step;

    // stage h into LDS (each thread 8 values, coalesced)
#pragma unroll
    for (int i = 0; i < 8; ++i)
      hs[threadIdx.x + i * 128] = h[threadIdx.x + i * 128];
    __syncthreads();

    // hgates[r] = dot(w_hh[r,:], h)   (bf16 weights, f32 accumulate)
    float acc = 0.0f;
#pragma unroll 4
    for (int k8 = 0; k8 < HH / 8; ++k8) {
      const uint4 w4 = wq[(long)k8 * G3 + r];
      const float4 h0 = *(const float4*)&hs[k8 * 8];
      const float4 h1 = *(const float4*)&hs[k8 * 8 + 4];
      acc += bflo(w4.x) * h0.x + bfhi(w4.x) * h0.y
           + bflo(w4.y) * h0.z + bfhi(w4.y) * h0.w
           + bflo(w4.z) * h1.x + bfhi(w4.z) * h1.y
           + bflo(w4.w) * h1.z + bfhi(w4.w) * h1.w;
    }
    hgd[r] = acc;

    grid_barrier(bar, NBLK);                 // hgates complete & visible

    if (r < HH) {
      const int j = r;
      const float hr = hgd[j];
      const float hz = hgd[HH + j];
      const float hn = hgd[2 * HH + j];
      const float* igt = ig + (long)t * G3;
      const float rr = 1.0f / (1.0f + __expf(-(igt[j] + hr)));
      const float zz = 1.0f / (1.0f + __expf(-(igt[HH + j] + hz)));
      const float nn = tanhf(igt[2 * HH + j] + bn[j] + rr * hn);
      const float hnew = (1.0f - zz) * nn + zz * hs[j];
      h[j] = hnew;
      const long o = (long)t * 2048 + dir * HH + j;
      xnext_bf[o] = f2bf(hnew);
      if (write_f32) out_f32[o] = hnew;
    }

    grid_barrier(bar, NBLK);                 // h update visible before next step
  }
}

// ---------------------------------------------------------------- launch
extern "C" void kernel_launch(void* const* d_in, const int* in_sizes, int n_in,
                              void* d_out, int out_size, void* d_ws, size_t ws_size,
                              hipStream_t stream) {
  (void)in_sizes; (void)n_in; (void)out_size; (void)ws_size;

  const float* xs = (const float*)d_in[0];
  // d_in layout: xs, then fwd_params[l]={w_ih,w_hh,b,bn} l=0..3, then bwd_params.
  auto prm = [&](int bwd, int l, int which) -> const float* {
    return (const float*)d_in[1 + bwd * 16 + l * 4 + which];
  };

  char* ws = (char*)d_ws;
  size_t off = 0;
  auto take = [&](size_t bytes) -> char* {
    char* p = ws + off;
    off = (off + bytes + 255) & ~(size_t)255;
    return p;
  };

  Bar*            bar  = (Bar*)           take(256);
  float*          hbuf = (float*)         take(2 * HH * sizeof(float));
  float*          hg   = (float*)         take(2 * G3 * sizeof(float));
  unsigned short* xbf  = (unsigned short*)take((size_t)TT * 2048 * 2);          // 16 MB
  float*          igF  = (float*)         take((size_t)TT * G3 * 4);            // 48 MB
  float*          igB  = (float*)         take((size_t)TT * G3 * 4);            // 48 MB
  unsigned short* wihF = (unsigned short*)take((size_t)G3 * 2048 * 2);          // 12.6 MB
  unsigned short* wihB = (unsigned short*)take((size_t)G3 * 2048 * 2);          // 12.6 MB
  uint4*          wqF  = (uint4*)         take((size_t)(HH / 8) * G3 * 16);     // 6.3 MB
  uint4*          wqB  = (uint4*)         take((size_t)(HH / 8) * G3 * 16);     // 6.3 MB

  init_bar_kernel<<<1, 64, 0, stream>>>(bar);

  // layer-0 input -> bf16 staging
  {
    long n = (long)TT * 1024;
    f32_to_bf16_kernel<<<(int)((n + 255) / 256), 256, 0, stream>>>(xs, xbf, n);
  }

  for (int l = 0; l < 4; ++l) {
    const int K = (l == 0) ? 1024 : 2048;
    const long wn = (long)G3 * K;

    f32_to_bf16_kernel<<<(int)((wn + 255) / 256), 256, 0, stream>>>(prm(0, l, 0), wihF, wn);
    f32_to_bf16_kernel<<<(int)((wn + 255) / 256), 256, 0, stream>>>(prm(1, l, 0), wihB, wn);

    const long pn = (long)(HH / 8) * G3;
    pack_whh_kernel<<<(int)((pn + 255) / 256), 256, 0, stream>>>(prm(0, l, 1), wqF);
    pack_whh_kernel<<<(int)((pn + 255) / 256), 256, 0, stream>>>(prm(1, l, 1), wqB);

    dim3 g(24, TT / 32, 2);   // 24 ntile-groups x 128 mtiles(32-row) x 2 directions
    wmma_igates_kernel<<<g, 256, 0, stream>>>(xbf, wihF, wihB,
                                              prm(0, l, 2), prm(1, l, 2),
                                              igF, igB, K);

    const int last = (l == 3);
    gru_scan_kernel<<<NBLK, 128, 0, stream>>>(igF, igB, wqF, wqB,
                                              prm(0, l, 3), prm(1, l, 3),
                                              xbf, (float*)d_out, last,
                                              hbuf, hg, bar);
  }
}
